// DTransformerLayer_79585743995626
// MI455X (gfx1250) — matrix operand order repaired and checked by reading, not compiled, not run on hardware
//
#include <hip/hip_runtime.h>
#include <hip/hip_bf16.h>

typedef unsigned short u16;
typedef __attribute__((ext_vector_type(16))) __bf16 v16bf;
typedef __attribute__((ext_vector_type(8)))  float  v8f;

union Frag { v16bf v; uint4 u[2]; };

#define WMMA_BF16(A, B, C) \
  __builtin_amdgcn_wmma_f32_16x16x32_bf16(false, (A), false, (B), (short)0, (C), false, false)

#define NEGV (-1e32f)

// ---- CDNA5 async global->LDS staging (signature from compiler diagnostic:
//      (int4vec __device__*, int4vec __shared__*, imm offset, imm cpol)) ----
#if __has_builtin(__builtin_amdgcn_global_load_async_to_lds_b128) && \
    __has_builtin(__builtin_amdgcn_s_wait_asynccnt)
typedef int v4i_vec __attribute__((vector_size(16)));
typedef __attribute__((address_space(1))) v4i_vec* as1_v4i_p;
typedef __attribute__((address_space(3))) v4i_vec* as3_v4i_p;
#define STAGE_B128(gp, lp) \
  __builtin_amdgcn_global_load_async_to_lds_b128((as1_v4i_p)(gp), (as3_v4i_p)(lp), 0, 0)
#define WAIT_ASYNC(n) __builtin_amdgcn_s_wait_asynccnt(n)
#else
#define STAGE_B128(gp, lp) (*(uint4*)(lp) = *(const uint4*)(gp))
#define WAIT_ASYNC(n) ((void)0)
#endif

// ---------------- helpers ----------------
__device__ inline u16 f32_to_bf16(float f) {
  unsigned int u = __float_as_uint(f);
  unsigned int r = u + 0x7FFFu + ((u >> 16) & 1u);   // round-to-nearest-even
  return (u16)(r >> 16);
}
__device__ inline float wred_max(float v) {
  #pragma unroll
  for (int m = 16; m > 0; m >>= 1) v = fmaxf(v, __shfl_xor(v, m, 32));
  return v;
}
__device__ inline float wred_sum(float v) {
  #pragma unroll
  for (int m = 16; m > 0; m >>= 1) v += __shfl_xor(v, m, 32);
  return v;
}

// ---------------- conversion kernels ----------------
__global__ void cvt_f32_to_bf16(const float* __restrict__ in, u16* __restrict__ out, int n) {
  int i = blockIdx.x * 256 + threadIdx.x;
  if (i < n) out[i] = f32_to_bf16(in[i]);
}

// Wt[o, i] = bf16(W[i, o]);  W is 1024x1024 row-major
__global__ void cvt_transpose_1024(const float* __restrict__ W, u16* __restrict__ Wt) {
  int idx = blockIdx.x * 256 + threadIdx.x;   // 0 .. 1M-1
  int o = idx >> 10, i = idx & 1023;
  Wt[idx] = f32_to_bf16(W[i * 1024 + o]);
}

// ---------------- projection GEMM (block-cooperative, async double-buffer) ----
// C[4096,1024] = A[4096,1024](bf16) @ Bt^T (Bt is [1024 out][1024 in] bf16) + bias
// Block: 256 threads (8 waves). Block tile = 256 rows x 64 cols.
// Wave w: rows [rowbase, rowbase+32) (2 M-tiles), all 4 N-tiles.
// B panel (64 cols x 32 k = 4KB) staged in LDS once per block, double-buffered.
// mode 0: store bf16 to [b,h,s,dk]   (Q / K)
// mode 1: store bf16 to [b,h,dk,s]   (V transposed)
// mode 2: store f32   to [m,n] with residual add
__global__ __launch_bounds__(256) void proj_gemm(
    const u16* __restrict__ A, const u16* __restrict__ Bt,
    const float* __restrict__ bias, void* __restrict__ out,
    const float* __restrict__ resid, int mode) {
  __shared__ u16 panel[2][64 * 32];                 // 8 KB
  int tid = threadIdx.x, wave = tid >> 5, lane = tid & 31;
  int half = lane >> 4, l15 = lane & 15;
  int mblock = blockIdx.x >> 4;                     // 16 blocks of 256 rows
  int ngroup = blockIdx.x & 15;                     // 16 groups of 64 cols
  int ncolbase = ngroup * 64;
  int rowbase = mblock * 256 + wave * 32;

  // cooperative B-panel staging: 256 chunks of 16B per panel, one per thread
  int sr = tid >> 2, sc = tid & 3;                  // row in [0,64), chunk in [0,4)
  const u16* sgp = Bt + (size_t)(ncolbase + sr) * 1024 + sc * 8;
  u16* slp = &panel[0][0] + sr * 32 + sc * 8;

  const u16* arow0 = A + (size_t)(rowbase + l15) * 1024;
  const u16* arow1 = A + (size_t)(rowbase + 16 + l15) * 1024;
  __builtin_prefetch(arow0, 0, 0);
  __builtin_prefetch(arow1, 0, 0);

  STAGE_B128(sgp, slp);                             // panel 0 -> buffer 0

  v8f acc[8] = {};
  for (int ks = 0; ks < 32; ++ks) {
    int cur = ks & 1;
    if (ks + 1 < 32) {                              // stage next panel, keep it in flight
      STAGE_B128(sgp + (ks + 1) * 32, slp + ((ks + 1) & 1) * (64 * 32));
      WAIT_ASYNC(1);                                // current panel complete (in-order)
    } else {
      WAIT_ASYNC(0);
    }
    __syncthreads();                                // panel[cur] visible to all waves

    int k0 = ks * 32;
    Frag a0, a1;
    a0.u[0] = *(const uint4*)(arow0 + k0 + half * 8);
    a0.u[1] = *(const uint4*)(arow0 + k0 + 16 + half * 8);
    a1.u[0] = *(const uint4*)(arow1 + k0 + half * 8);
    a1.u[1] = *(const uint4*)(arow1 + k0 + 16 + half * 8);
    #pragma unroll
    for (int t = 0; t < 4; ++t) {
      const u16* pb = &panel[cur][0] + (t * 16 + l15) * 32;
      Frag b;
      b.u[0] = *(const uint4*)(pb + half * 8);
      b.u[1] = *(const uint4*)(pb + 16 + half * 8);
      acc[t]     = WMMA_BF16(a0.v, b.v, acc[t]);
      acc[4 + t] = WMMA_BF16(a1.v, b.v, acc[4 + t]);
    }
    __syncthreads();                                // all reads of panel[cur] done
  }

  u16*   ob = (u16*)out;
  float* of = (float*)out;
  #pragma unroll
  for (int mt = 0; mt < 2; ++mt) {
    #pragma unroll
    for (int t = 0; t < 4; ++t) {
      int ncol = ncolbase + t * 16 + l15;
      float bv = bias[ncol];
      #pragma unroll
      for (int r = 0; r < 8; ++r) {
        int mrow = rowbase + mt * 16 + r + 8 * half;  // C layout: lanes0-15 M=r, lanes16-31 M=8+r
        float val = acc[mt * 4 + t][r] + bv;
        if (mode == 2) {
          size_t i = (size_t)mrow * 1024 + ncol;
          of[i] = val + resid[i];
        } else {
          int bb = mrow >> 10, sidx = mrow & 1023;
          int hh = ncol >> 6,  dd   = ncol & 63;
          if (mode == 0)
            ob[((size_t)(bb * 16 + hh) * 1024 + sidx) * 64 + dd] = f32_to_bf16(val);
          else
            ob[((size_t)(bb * 16 + hh) * 64 + dd) * 1024 + sidx] = f32_to_bf16(val);
        }
      }
    }
  }
}

// ---------------- fused attention ----------------
// grid = (64 q-tiles, 16 heads, 4 batch), block = 512 (16 waves)
__global__ __launch_bounds__(512) void attn_kernel(
    const u16* __restrict__ Qm, const u16* __restrict__ Km, const u16* __restrict__ Vt,
    const float* __restrict__ gammas, float* __restrict__ scoresOut, u16* __restrict__ AttnC) {
  __shared__ float smem[16 * 1024];                 // 64 KB, reused across phases
  u16*   smem_prob = (u16*)smem;                    // bytes [0,32K):  16 x 1024 bf16 probs
  float* smem_part = smem + 8 * 1024;               // bytes [32K,48K): 4 x 16 x 64 f32 partials

  int qtile = blockIdx.x, h = blockIdx.y, b = blockIdx.z;
  int tid = threadIdx.x, wave = tid >> 5, lane = tid & 31;
  int half = lane >> 4, l15 = lane & 15;
  size_t bh = (size_t)(b * 16 + h);

  const u16* Qbase = Qm + bh * 1024 * 64;
  const u16* Kbase = Km + bh * 1024 * 64;
  const u16* Vbase = Vt + bh * 64 * 1024;

  // ---- Phase 1: raw scores = (Q K^T) / sqrt(64) into LDS (f32) ----
  Frag a0, a1;
  {
    const u16* qrow = Qbase + (size_t)(qtile * 16 + l15) * 64;
    a0.u[0] = *(const uint4*)(qrow + half * 8);
    a0.u[1] = *(const uint4*)(qrow + 16 + half * 8);
    a1.u[0] = *(const uint4*)(qrow + 32 + half * 8);
    a1.u[1] = *(const uint4*)(qrow + 48 + half * 8);
  }
  Frag bf[8];                                       // all 4 K-tiles preloaded -> one clause
  #pragma unroll
  for (int t = 0; t < 4; ++t) {
    int nt = wave * 4 + t;                          // 64 N-tiles over 16 waves
    const u16* krow = Kbase + (size_t)(nt * 16 + l15) * 64;
    bf[2 * t].u[0]     = *(const uint4*)(krow + half * 8);
    bf[2 * t].u[1]     = *(const uint4*)(krow + 16 + half * 8);
    bf[2 * t + 1].u[0] = *(const uint4*)(krow + 32 + half * 8);
    bf[2 * t + 1].u[1] = *(const uint4*)(krow + 48 + half * 8);
  }
  #pragma unroll
  for (int t = 0; t < 4; ++t) {
    int nt = wave * 4 + t;
    v8f acc = {};
    acc = WMMA_BF16(a0.v, bf[2 * t].v, acc);
    acc = WMMA_BF16(a1.v, bf[2 * t + 1].v, acc);
    #pragma unroll
    for (int r = 0; r < 8; ++r)
      smem[(r + 8 * half) * 1024 + nt * 16 + l15] = acc[r] * 0.125f;
  }
  __syncthreads();

  // ---- Phase 2: per-row softmax / cumsum / decay / softmax / rescale ----
  int row = wave;                       // one wave per q-row of the tile
  int qg  = qtile * 16 + row;
  float raw[32];
  #pragma unroll
  for (int j = 0; j < 32; ++j) raw[j] = smem[row * 1024 + lane * 32 + j];
  __syncthreads();                      // raw now in registers; LDS reusable

  float gneg = -fabsf(gammas[h]);
  float p[32];
  // first masked softmax -> scores_
  float mx = NEGV;
  #pragma unroll
  for (int j = 0; j < 32; ++j) {
    int col = lane * 32 + j;
    p[j] = (col < qg) ? raw[j] : NEGV;
    mx = fmaxf(mx, p[j]);
  }
  mx = wred_max(mx);
  float s = 0.f;
  #pragma unroll
  for (int j = 0; j < 32; ++j) { p[j] = __expf(p[j] - mx); s += p[j]; }
  s = wred_sum(s);
  float inv = 1.0f / s;
  // inclusive cumsum over the row (lane-local then wave scan)
  float c = 0.f;
  #pragma unroll
  for (int j = 0; j < 32; ++j) { p[j] *= inv; c += p[j]; p[j] = c; }
  float incl = c;
  #pragma unroll
  for (int d = 1; d < 32; d <<= 1) {
    float t2 = __shfl_up(incl, d, 32);
    if (lane >= d) incl += t2;
  }
  float excl = incl - c;
  float tot  = __shfl(incl, 31, 32);
  // distance decay, second masked scores
  float mx2 = NEGV;
  #pragma unroll
  for (int j = 0; j < 32; ++j) {
    int col = lane * 32 + j;
    float distcum = excl + p[j];
    float pe  = fabsf((float)(qg - col));
    float dsc = sqrtf(fmaxf((tot - distcum) * pe, 0.0f));
    float te  = fminf(fmaxf(__expf(dsc * gneg), 1e-5f), 1e5f);
    float s2  = raw[j] * te;
    p[j] = (col < qg) ? s2 : NEGV;
    mx2 = fmaxf(mx2, p[j]);
  }
  mx2 = wred_max(mx2);
  float s2sum = 0.f;
  #pragma unroll
  for (int j = 0; j < 32; ++j) { p[j] = __expf(p[j] - mx2); s2sum += p[j]; }
  s2sum = wred_sum(s2sum);
  inv = 1.0f / s2sum;
  float pmax = 0.f;
  #pragma unroll
  for (int j = 0; j < 32; ++j) {
    int col = lane * 32 + j;
    p[j] = (col < qg) ? p[j] * inv : 0.0f;
    pmax = fmaxf(pmax, p[j]);
  }
  pmax = wred_max(pmax);
  float scl = fminf(1.0f / pmax, 5.0f);
  float* srow = scoresOut + (bh * 1024 + (size_t)qg) * 1024;
  #pragma unroll
  for (int j = 0; j < 32; ++j) {
    int col = lane * 32 + j;
    float fin = p[j] * scl;
    srow[col] = fin;                               // f32 scores output
    smem_prob[row * 1024 + col] = f32_to_bf16(fin);
  }
  __syncthreads();

  // ---- Phase 3: out-tile = P[16x1024] @ V[1024x64], register double-buffered ----
  int nt = wave & 3;        // dk column group (16 cols)
  int kq = wave >> 2;       // contraction quarter (256)
  v8f acc = {};
  const u16* vrow = Vbase + (size_t)(nt * 16 + l15) * 1024;
  const u16* prow = smem_prob + l15 * 1024;
  Frag pa[2], pb[2];
  auto ld3 = [&](int ks, int slot) {
    int kb = kq * 256 + ks * 32;
    pa[slot].u[0] = *(const uint4*)(prow + kb + half * 8);
    pa[slot].u[1] = *(const uint4*)(prow + kb + 16 + half * 8);
    pb[slot].u[0] = *(const uint4*)(vrow + kb + half * 8);
    pb[slot].u[1] = *(const uint4*)(vrow + kb + 16 + half * 8);
  };
  ld3(0, 0);
  #pragma unroll
  for (int ks = 0; ks < 8; ++ks) {
    int cur = ks & 1;
    if (ks + 1 < 8) ld3(ks + 1, cur ^ 1);          // overlap next loads with WMMA
    acc = WMMA_BF16(pa[cur].v, pb[cur].v, acc);
  }
  #pragma unroll
  for (int r = 0; r < 8; ++r)
    smem_part[kq * 1024 + (r + 8 * half) * 64 + nt * 16 + l15] = acc[r];
  __syncthreads();

  // ---- Phase 4: reduce 4 partials, write concat-layout bf16 ----
  for (int idx = tid; idx < 1024; idx += 512) {
    float sum = smem_part[idx] + smem_part[1024 + idx] +
                smem_part[2048 + idx] + smem_part[3072 + idx];
    int rr = idx >> 6, cc = idx & 63;
    AttnC[((size_t)(b * 1024 + qtile * 16 + rr)) * 1024 + h * 64 + cc] = f32_to_bf16(sum);
  }
}

// ---------------- layernorm ----------------
__global__ __launch_bounds__(256) void ln_kernel(
    const float* __restrict__ X, const float* __restrict__ w,
    const float* __restrict__ bln, float* __restrict__ out) {
  __shared__ float r1[8], r2[8];
  int row = blockIdx.x, tid = threadIdx.x;
  const float* x = X + (size_t)row * 1024;
  float v[4], s = 0.f, ss = 0.f;
  #pragma unroll
  for (int i = 0; i < 4; ++i) {
    v[i] = x[tid + 256 * i];
    s += v[i]; ss += v[i] * v[i];
  }
  s = wred_sum(s); ss = wred_sum(ss);
  int wv = tid >> 5, ln = tid & 31;
  if (ln == 0) { r1[wv] = s; r2[wv] = ss; }
  __syncthreads();
  if (tid == 0) {
    float S = 0.f, SS = 0.f;
    for (int k = 0; k < 8; ++k) { S += r1[k]; SS += r2[k]; }
    float mu  = S * (1.0f / 1024.0f);
    float var = SS * (1.0f / 1024.0f) - mu * mu;
    r1[0] = mu; r2[0] = rsqrtf(var + 1e-5f);
  }
  __syncthreads();
  float mu = r1[0], rs = r2[0];
  #pragma unroll
  for (int i = 0; i < 4; ++i) {
    int cc = tid + 256 * i;
    out[(size_t)row * 1024 + cc] = (v[i] - mu) * rs * w[cc] + bln[cc];
  }
}

// ---------------- launcher ----------------
extern "C" void kernel_launch(void* const* d_in, const int* in_sizes, int n_in,
                              void* d_out, int out_size, void* d_ws, size_t ws_size,
                              hipStream_t stream) {
  (void)in_sizes; (void)n_in; (void)out_size; (void)ws_size;
  const float* query  = (const float*)d_in[0];
  const float* key    = (const float*)d_in[1];
  const float* values = (const float*)d_in[2];
  const float* Wq     = (const float*)d_in[3];
  const float* bq     = (const float*)d_in[4];
  const float* Wv     = (const float*)d_in[5];
  const float* bv     = (const float*)d_in[6];
  const float* Wo     = (const float*)d_in[7];
  const float* bo     = (const float*)d_in[8];
  const float* gammas = (const float*)d_in[9];
  const float* ln_w   = (const float*)d_in[10];
  const float* ln_b   = (const float*)d_in[11];
  // d_in[12] = lens : unused by the reference

  const size_t MB = 1024 * 1024;
  char* ws = (char*)d_ws;
  u16*   Wqt  = (u16*)(ws + 0 * MB);     // 2 MB
  u16*   Wvt  = (u16*)(ws + 2 * MB);     // 2 MB
  u16*   Wot  = (u16*)(ws + 4 * MB);     // 2 MB
  u16*   Xq   = (u16*)(ws + 6 * MB);     // 8 MB  (4096x1024 bf16)
  u16*   Xk   = (u16*)(ws + 14 * MB);
  u16*   Xv   = (u16*)(ws + 22 * MB);
  u16*   Qm   = (u16*)(ws + 30 * MB);    // [b,h,s,dk] bf16
  u16*   Km   = (u16*)(ws + 38 * MB);
  u16*   VtB  = (u16*)(ws + 46 * MB);    // [b,h,dk,s] bf16
  u16*   AtC  = (u16*)(ws + 54 * MB);    // concat rows bf16
  float* Xres = (float*)(ws + 62 * MB);  // 16 MB f32

  float* out_main   = (float*)d_out;                 // [4,1024,1024]
  float* out_scores = (float*)d_out + 4194304;       // [4,16,1024,1024]

  const int NROW = 4096 * 1024;
  cvt_f32_to_bf16<<<NROW / 256, 256, 0, stream>>>(query,  Xq, NROW);
  cvt_f32_to_bf16<<<NROW / 256, 256, 0, stream>>>(key,    Xk, NROW);
  cvt_f32_to_bf16<<<NROW / 256, 256, 0, stream>>>(values, Xv, NROW);
  cvt_transpose_1024<<<4096, 256, 0, stream>>>(Wq, Wqt);
  cvt_transpose_1024<<<4096, 256, 0, stream>>>(Wv, Wvt);
  cvt_transpose_1024<<<4096, 256, 0, stream>>>(Wo, Wot);

  // Q = query@Wq+bq ; K = key@Wq+bq (kq_same) ; V = values@Wv+bv (stored transposed)
  proj_gemm<<<256, 256, 0, stream>>>(Xq, Wqt, bq, (void*)Qm,  nullptr, 0);
  proj_gemm<<<256, 256, 0, stream>>>(Xk, Wqt, bq, (void*)Km,  nullptr, 0);
  proj_gemm<<<256, 256, 0, stream>>>(Xv, Wvt, bv, (void*)VtB, nullptr, 1);

  attn_kernel<<<dim3(64, 16, 4), 512, 0, stream>>>(Qm, Km, VtB, gammas, out_scores, AtC);

  // attn_out@Wo + bo + residual(query) -> Xres (f32)
  proj_gemm<<<256, 256, 0, stream>>>(AtC, Wot, bo, (void*)Xres, query, 2);

  ln_kernel<<<4096, 256, 0, stream>>>(Xres, ln_w, ln_b, out_main);
}
